// Attention_70557722739394
// MI455X (gfx1250) — matrix operand order repaired and checked by reading, not compiled, not run on hardware
//
#include <hip/hip_runtime.h>
#include <math.h>

// ---------------------------------------------------------------------------
// Causal multi-head attention forward for MI455X (gfx1250), wave32 + WMMA.
// Matmuls via v_wmma_f32_16x16x32_bf16; GEMM weight tiles staged in LDS via
// async Global->LDS DMA (ASYNCcnt), double-buffered.
// ---------------------------------------------------------------------------

typedef __attribute__((ext_vector_type(4)))  __bf16 bf16x4;
typedef __attribute__((ext_vector_type(8)))  __bf16 bf16x8;
typedef __attribute__((ext_vector_type(16))) __bf16 bf16x16;
typedef __attribute__((ext_vector_type(8)))  float  f32x8;

constexpr int Bb = 2;      // batch
constexpr int Ss = 2048;   // sequence (power of two)
constexpr int Cc = 2048;   // channels
constexpr int Hh = 16;     // heads
constexpr int Dd = 128;    // head dim
constexpr int Mm = Bb * Ss;        // GEMM M = 4096
constexpr int SsShift = 11;        // log2(Ss)

// ---- WMMA fragment helpers ------------------------------------------------
// A fragment 16x32 (this code's contiguous k order, not ISA table order): lane = (li, h); row M = li;
// element e -> K = e + 16*h (matches B).  => one contiguous 32B load.
__device__ __forceinline__ bf16x16 load_a_frag(const __bf16* row, int k0, int h) {
  return *reinterpret_cast<const bf16x16*>(row + k0 + 16 * h);
}
// B fragment 32x16 (same contiguous k order as A, not ISA table order): lane (li, h)
// holds column N = li, elements K = e + 16*h.  => one contiguous 32B load.
__device__ __forceinline__ bf16x16 load_b_frag(const __bf16* row, int k0, int h) {
  return *reinterpret_cast<const bf16x16*>(row + k0 + 16 * h);
}
__device__ __forceinline__ f32x8 wmma_bf16(bf16x16 a, bf16x16 b, f32x8 c) {
  return __builtin_amdgcn_wmma_f32_16x16x32_bf16(false, a, false, b,
                                                 (short)0, c, false, false);
}

// ---- async Global -> LDS DMA (GLOBAL_LOAD_ASYNC_TO_LDS_B128, ASYNCcnt) ----
__device__ __forceinline__ void async_copy_b128(uint32_t lds_byte_off,
                                                const void* gaddr) {
  asm volatile("global_load_async_to_lds_b128 %0, %1, off"
               :: "v"(lds_byte_off), "v"(gaddr)
               : "memory");
}
__device__ __forceinline__ void wait_asynccnt0() {
  asm volatile("s_wait_asynccnt 0x0" ::: "memory");
}

// ---- fp32 -> bf16 conversion ----------------------------------------------
__global__ __launch_bounds__(256)
void cvt_f32_bf16(const float* __restrict__ in, __bf16* __restrict__ out, int n4) {
  int i = blockIdx.x * blockDim.x + threadIdx.x;
  if (i >= n4) return;
  const float4 v = reinterpret_cast<const float4*>(in)[i];
  bf16x4 o;
  o[0] = (__bf16)v.x; o[1] = (__bf16)v.y; o[2] = (__bf16)v.z; o[3] = (__bf16)v.w;
  reinterpret_cast<bf16x4*>(out)[i] = o;
}

// ---- bf16 NT-GEMM:  out[m,n] = (sum_k A[m,k]*W[n,k] + bias[n])*scale
// Fixed geometry: M = Bb*Ss = 4096, N = K = Cc = 2048.
// Block: 256 thr (8 waves, 2x4), tile 64(M) x 256(N).  B (weight) k-slices are
// DMA'd into a double-buffered LDS tile (async-to-LDS), A comes from global.
// MODE 0: store bf16 row-major (M,N)
// MODE 1: store bf16 transposed per batch: out[(m>>11)*Cc + n][m&2047] (V^T)
// MODE 2: store fp32 row-major (final projection)
template <int MODE>
__global__ __launch_bounds__(256)
void gemm_wmma(const __bf16* __restrict__ A, const __bf16* __restrict__ W,
               const float* __restrict__ bias, void* __restrict__ outp,
               float scale) {
  constexpr int K    = Cc;       // reduction length
  constexpr int N    = Cc;       // output columns
  constexpr int ROWS = 256;      // weight rows staged per block
  constexpr int RSTR = 40;       // bf16 per LDS row (80B: 64B data + 16B pad)
  __shared__ __align__(16) __bf16 smemB[2][ROWS * RSTR];

  const int tid  = threadIdx.x;
  const int lane = tid & 31;
  const int wave = tid >> 5;
  const int wm = wave >> 2;            // 0..1
  const int wn = wave & 3;             // 0..3
  const int li = lane & 15;
  const int h  = lane >> 4;
  const int mBase  = blockIdx.y * 64 + wm * 32;
  const int nBlock = blockIdx.x * 256;
  const int nBase  = nBlock + wn * 64;

  const __bf16* arow[2];
#pragma unroll
  for (int i = 0; i < 2; ++i) arow[i] = A + (size_t)(mBase + 16 * i + li) * K;

  // Per-thread DMA source: one weight row per thread; 64B (4x b128) per k-step.
  const __bf16* wsrc = W + (size_t)(nBlock + tid) * K;
  const uint32_t ldsRow[2] = {
      (uint32_t)(uintptr_t)&smemB[0][tid * RSTR],
      (uint32_t)(uintptr_t)&smemB[1][tid * RSTR]};

  float bv[4];
#pragma unroll
  for (int j = 0; j < 4; ++j) bv[j] = bias[nBase + 16 * j + li];

  // Prologue: DMA first k-slice into buffer 0.
#pragma unroll
  for (int c = 0; c < 4; ++c)
    async_copy_b128(ldsRow[0] + c * 16, wsrc + c * 8);
  wait_asynccnt0();
  __syncthreads();

  f32x8 acc[2][4] = {};
  for (int k0 = 0; k0 < K; k0 += 32) {
    const int  cur  = (k0 >> 5) & 1;
    const bool more = (k0 + 32) < K;
    if (more) {                         // DMA next slice into other buffer
#pragma unroll
      for (int c = 0; c < 4; ++c)
        async_copy_b128(ldsRow[cur ^ 1] + c * 16, wsrc + (k0 + 32) + c * 8);
    }
    bf16x16 af[2];
#pragma unroll
    for (int i = 0; i < 2; ++i) {
      af[i] = load_a_frag(arow[i], k0, h);
      __builtin_prefetch(arow[i] + k0 + 256, 0, 3);
    }
    const __bf16* bbuf = &smemB[cur][0];
    bf16x16 bfr[4];
#pragma unroll
    for (int j = 0; j < 4; ++j) {
      const int r = wn * 64 + 16 * j + li;   // local weight row
      bfr[j] = *reinterpret_cast<const bf16x16*>(bbuf + r * RSTR + 16 * h);
    }
#pragma unroll
    for (int i = 0; i < 2; ++i)
#pragma unroll
      for (int j = 0; j < 4; ++j)
        acc[i][j] = wmma_bf16(af[i], bfr[j], acc[i][j]);
    if (more) {                         // next buffer ready + all reads done
      wait_asynccnt0();
      __syncthreads();
    }
  }

  // Epilogue. D layout: lane holds column n = li, rows m = v + 8*h.
#pragma unroll
  for (int j = 0; j < 4; ++j) {
    const int n = nBase + 16 * j + li;
#pragma unroll
    for (int i = 0; i < 2; ++i) {
#pragma unroll
      for (int v = 0; v < 8; ++v) {
        const int m = mBase + 16 * i + v + 8 * h;
        const float val = (acc[i][j][v] + bv[j]) * scale;
        if constexpr (MODE == 0) {
          ((__bf16*)outp)[(size_t)m * N + n] = (__bf16)val;
        } else if constexpr (MODE == 1) {
          const int b = m >> SsShift, s2 = m & (Ss - 1);
          ((__bf16*)outp)[((size_t)b * N + n) * Ss + s2] = (__bf16)val;
        } else {
          ((float*)outp)[(size_t)m * N + n] = val;
        }
      }
    }
  }
}

// ---- flash attention (transposed formulation) -----------------------------
// Per wave: 16 query rows of one (b, head). Online softmax over key blocks
// of 32.  S^T = K·Q^T keeps each query's scores in-lane (q = lane&15);
// P^T (the D-fragments, exponentiated) maps directly onto the B operand of
// O^T += V^T·P^T after one cross-half shfl_xor(16) exchange.
__global__ __launch_bounds__(128)
void flash_attn(const __bf16* __restrict__ Qb, const __bf16* __restrict__ Kb,
                const __bf16* __restrict__ Vtb, __bf16* __restrict__ Ob) {
  const int lane = threadIdx.x & 31;
  const int wave = threadIdx.x >> 5;     // 0..3
  const int li = lane & 15;
  const int hh = lane >> 4;
  const int b    = blockIdx.z;
  const int head = blockIdx.y;
  const int q0 = blockIdx.x * 64 + wave * 16;
  const int qs = q0 + li;                 // this lane's query row (as B column)
  const int colBase = head * Dd;

  // Q^T B-fragments, whole head dim (4 x 32), kept in registers. Q pre-scaled.
  const __bf16* qrow = Qb + ((size_t)b * Ss + qs) * Cc + colBase;
  bf16x16 qf[4];
#pragma unroll
  for (int c = 0; c < 4; ++c) qf[c] = load_b_frag(qrow, 32 * c, hh);

  f32x8 ot[8] = {};                       // O^T: lane col q=li, row d = v+8h (+16t)
  const float NEG_INF = -__builtin_inff();
  float m_run = NEG_INF, l_run = 0.0f;

  const int kmax = q0 + 15;               // causal: keys <= max query row in strip
  for (int k0 = 0; k0 <= kmax; k0 += 32) {
    // ---- S^T = K · Q^T  (two 16-key sub-tiles) ----
    f32x8 st[2];
#pragma unroll
    for (int kt = 0; kt < 2; ++kt) {
      const __bf16* krow =
          Kb + ((size_t)b * Ss + k0 + 16 * kt + li) * Cc + colBase;
      f32x8 sacc = {};
#pragma unroll
      for (int c = 0; c < 4; ++c)
        sacc = wmma_bf16(load_a_frag(krow, 32 * c, hh), qf[c], sacc);
      st[kt] = sacc;
    }

    // ---- causal mask + online softmax (row = fixed q = li, in-lane) ----
    float p0[8], p1[8];
    float mx = NEG_INF;
#pragma unroll
    for (int v = 0; v < 8; ++v) {
      const int key0 = k0 + v + 8 * hh;
      const int key1 = k0 + 16 + v + 8 * hh;
      const float s0 = (key0 <= qs) ? st[0][v] : NEG_INF;
      const float s1 = (key1 <= qs) ? st[1][v] : NEG_INF;
      st[0][v] = s0; st[1][v] = s1;
      mx = fmaxf(mx, fmaxf(s0, s1));
    }
    mx = fmaxf(mx, __shfl_xor(mx, 16));            // other 16 keys live in lane^16
    const float m_new = fmaxf(m_run, mx);
    const float alpha = __expf(m_run - m_new);     // exp(-inf)=0 on first block
    float ps = 0.0f;
#pragma unroll
    for (int v = 0; v < 8; ++v) {
      p0[v] = __expf(st[0][v] - m_new);
      p1[v] = __expf(st[1][v] - m_new);
      ps += p0[v] + p1[v];
    }
    ps += __shfl_xor(ps, 16);
    l_run = l_run * alpha + ps;
    m_run = m_new;
#pragma unroll
    for (int t = 0; t < 8; ++t)
#pragma unroll
      for (int v = 0; v < 8; ++v) ot[t][v] *= alpha;

    // ---- pack P^T into the B-fragment layout (keys K = e + 16h) ----
    // lane (q,h=0) owns keys {v, 16+v}; lane (q,h=1) owns {8+v, 24+v}.
    float x0[8], x1[8];
#pragma unroll
    for (int v = 0; v < 8; ++v) {
      x0[v] = __shfl_xor(p0[v], 16);
      x1[v] = __shfl_xor(p1[v], 16);
    }
    bf16x16 pf;
#pragma unroll
    for (int e = 0; e < 8; ++e) {
      pf[e]     = (__bf16)(hh ? x1[e] : p0[e]);    // keys 16h + 0..7
      pf[e + 8] = (__bf16)(hh ? p1[e] : x0[e]);    // keys 16h + 8..15
    }

    // ---- O^T += V^T · P^T  (V^T rows contiguous in Vtb = (B, C, S)) ----
#pragma unroll
    for (int t = 0; t < 8; ++t) {
      const __bf16* vrow =
          Vtb + ((size_t)b * Cc + colBase + 16 * t + li) * Ss + k0;
      ot[t] = wmma_bf16(load_a_frag(vrow, 0, hh), pf, ot[t]);
    }
  }

  // ---- normalize and store merged-head output (bf16, (B*S, C)) ----
  const float invl = 1.0f / l_run;
  __bf16* orow = Ob + ((size_t)b * Ss + qs) * Cc + colBase;
#pragma unroll
  for (int t = 0; t < 8; ++t)
#pragma unroll
    for (int v = 0; v < 8; ++v)
      orow[16 * t + v + 8 * hh] = (__bf16)(ot[t][v] * invl);
}

// ---------------------------------------------------------------------------
extern "C" void kernel_launch(void* const* d_in, const int* in_sizes, int n_in,
                              void* d_out, int out_size, void* d_ws, size_t ws_size,
                              hipStream_t stream) {
  (void)in_sizes; (void)n_in; (void)out_size; (void)ws_size;
  const float* x  = (const float*)d_in[0];
  const float* Wq = (const float*)d_in[1];
  const float* bq = (const float*)d_in[2];
  const float* Wk = (const float*)d_in[3];
  const float* bk = (const float*)d_in[4];
  const float* Wv = (const float*)d_in[5];
  const float* bv = (const float*)d_in[6];
  const float* Wo = (const float*)d_in[7];
  const float* bo = (const float*)d_in[8];
  // d_in[9] = mask: causal mask is applied analytically in flash_attn.
  float* out = (float*)d_out;

  // Workspace carve-up (~112 MB of bf16 staging).
  char* ws = (char*)d_ws;
  const size_t NX = (size_t)Bb * Ss * Cc;   // 8,388,608 activations
  const size_t NW = (size_t)Cc * Cc;        // 4,194,304 weights
  __bf16* xb  = (__bf16*)ws; ws += NX * 2;
  __bf16* Wqb = (__bf16*)ws; ws += NW * 2;
  __bf16* Wkb = (__bf16*)ws; ws += NW * 2;
  __bf16* Wvb = (__bf16*)ws; ws += NW * 2;
  __bf16* Wob = (__bf16*)ws; ws += NW * 2;
  __bf16* Qb  = (__bf16*)ws; ws += NX * 2;  // pre-scaled by 1/sqrt(D)
  __bf16* Kb  = (__bf16*)ws; ws += NX * 2;
  __bf16* Vtb = (__bf16*)ws; ws += NX * 2;  // (B, C, S) transposed
  __bf16* Ob  = (__bf16*)ws; ws += NX * 2;  // attention output, merged heads

  // 1) fp32 -> bf16 staging
  cvt_f32_bf16<<<(int)(NX / 4 / 256), 256, 0, stream>>>(x,  xb,  (int)(NX / 4));
  cvt_f32_bf16<<<(int)(NW / 4 / 256), 256, 0, stream>>>(Wq, Wqb, (int)(NW / 4));
  cvt_f32_bf16<<<(int)(NW / 4 / 256), 256, 0, stream>>>(Wk, Wkb, (int)(NW / 4));
  cvt_f32_bf16<<<(int)(NW / 4 / 256), 256, 0, stream>>>(Wv, Wvb, (int)(NW / 4));
  cvt_f32_bf16<<<(int)(NW / 4 / 256), 256, 0, stream>>>(Wo, Wob, (int)(NW / 4));

  // 2) QKV projections (M=4096, N=K=2048). Q scaled by 1/sqrt(128).
  const dim3 ggrid(Cc / 256, Mm / 64);
  const float qscale = 0.08838834764831845f;  // 1/sqrt(128)
  gemm_wmma<0><<<ggrid, 256, 0, stream>>>(xb, Wqb, bq, Qb, qscale);
  gemm_wmma<0><<<ggrid, 256, 0, stream>>>(xb, Wkb, bk, Kb, 1.0f);
  gemm_wmma<1><<<ggrid, 256, 0, stream>>>(xb, Wvb, bv, Vtb, 1.0f);

  // 3) causal flash attention: one wave per 16 query rows per (b, head)
  flash_attn<<<dim3(Ss / 64, Hh, Bb), 128, 0, stream>>>(Qb, Kb, Vtb, Ob);

  // 4) output projection -> fp32 d_out
  gemm_wmma<2><<<ggrid, 256, 0, stream>>>(Ob, Wob, bo, out, 1.0f);
}
